// HFullyConnectedTensorProduct_40896678592923
// MI455X (gfx1250) — compile-verified, matrix operand order checked by reading, not run on hardware
//
#include <hip/hip_runtime.h>

typedef __attribute__((ext_vector_type(16))) __bf16 v16bf;
typedef __attribute__((ext_vector_type(8)))  float  v8f;
typedef __attribute__((ext_vector_type(4)))  unsigned int v4u;

#define B_DWORDS 81920   // 5 tensors * 2 w-halves * 32 kb * 256 dwords = 320 KB

// ---------------- bf16 packed helpers ----------------

// round-to-nearest-even f32 -> bf16 pair packer (pure integer; prep path)
__device__ __forceinline__ unsigned int pack_bf16(float a, float b) {
    unsigned int ua = __float_as_uint(a);
    unsigned int ub = __float_as_uint(b);
    unsigned int ra = (ua + 0x7FFFu + ((ua >> 16) & 1u)) >> 16;
    unsigned int rb = (ub + 0x7FFFu + ((ub >> 16) & 1u)) >> 16;
    return (ra & 0xFFFFu) | (rb << 16);
}

// hardware cvt: d = {bf16(a), bf16(b)}  (mnemonic verified in compiler output)
__device__ __forceinline__ unsigned int cvt_pk(float a, float b) {
    unsigned int d;
    asm("v_cvt_pk_bf16_f32 %0, %1, %2" : "=v"(d) : "v"(a), "v"(b));
    return d;
}

// replicated bf16 scalar pair
__device__ __forceinline__ unsigned int splat_bf16(float s) {
    unsigned int d;
    asm("v_cvt_pk_bf16_f32 %0, %1, %1" : "=v"(d) : "v"(s));
    return d;
}

// packed bf16 multiply / fma (CDNA5 VOP3P)
__device__ __forceinline__ unsigned int pk_mul_bf16(unsigned int a, unsigned int b) {
    unsigned int d;
    asm("v_pk_mul_bf16 %0, %1, %2" : "=v"(d) : "v"(a), "v"(b));
    return d;
}
__device__ __forceinline__ unsigned int pk_fma_bf16(unsigned int a, unsigned int b,
                                                    unsigned int c) {
    unsigned int d;
    asm("v_pk_fma_bf16 %0, %1, %2, %3" : "=v"(d) : "v"(a), "v"(b), "v"(c));
    return d;
}

// ---------------- fragments ----------------

struct Frag { unsigned int w[8]; };   // 16 bf16 = one WMMA A/B operand
struct U8 { v4u a, b; };

__device__ __forceinline__ v16bf load_frag_lds(const unsigned int* p) {
    U8 t;
    t.a = *(const v4u*)(p);
    t.b = *(const v4u*)(p + 4);
    return __builtin_bit_cast(v16bf, t);
}

__device__ __forceinline__ v8f wmma_bf16(v16bf a, v16bf b, v8f c) {
    return __builtin_amdgcn_wmma_f32_16x16x32_bf16(false, a, false, b, (short)0, c,
                                                   false, false);
}

// A = base * s  (8 v_pk_mul_bf16)
__device__ __forceinline__ v16bf mkfrag(const Frag& base, unsigned int spair) {
    Frag r;
#pragma unroll
    for (int i = 0; i < 8; ++i) r.w[i] = pk_mul_bf16(base.w[i], spair);
    return __builtin_bit_cast(v16bf, r);
}

// A = fa * sa - fb * sb  (sb passed pre-negated; 8 mul + 8 fma)
__device__ __forceinline__ v16bf mkfrag2(const Frag& fa, unsigned int sa,
                                         const Frag& fb, unsigned int neg_sb) {
    Frag r;
#pragma unroll
    for (int i = 0; i < 8; ++i)
        r.w[i] = pk_fma_bf16(fb.w[i], neg_sb, pk_mul_bf16(fa.w[i], sa));
    return __builtin_bit_cast(v16bf, r);
}

// ---------------- weight prep ----------------
// dword index = ((t*2 + h)*32 + kb)*256 + lane*8 + p
// t: 0=w000,1=w110,2=w011,3=w101,4=w111 ; h = 16-wide w-half
__device__ __forceinline__ unsigned int weight_dword(
        int idx, const float* w000, const float* w011, const float* w101,
        const float* w110, const float* w111)
{
    const float A_COM = 0.0220970869120796f;   // 1/sqrt(2048) = A000=A011=A101=A111
    const float A_110 = 0.0127585411087784f;   // 1/sqrt(6144)
    const int th   = idx >> 13;
    const int t    = th >> 1;
    const int h    = th & 1;
    const int r    = idx & 8191;
    const int kb   = r >> 8;         // K-chunk == u index
    const int r3   = r & 255;
    const int ln   = r3 >> 3;
    const int p    = r3 & 7;
    const int half = ln >> 4;
    const int w    = h * 16 + (ln & 15);
    const int v0   = ((p < 4) ? 0 : 16) + (half ? 8 : 0) + 2 * (p & 3);
    const float* W; float sc;
    switch (t) {
        case 0:  W = w000; sc = A_COM; break;
        case 1:  W = w110; sc = A_110; break;
        case 2:  W = w011; sc = A_COM; break;
        case 3:  W = w101; sc = A_COM; break;
        default: W = w111; sc = A_COM; break;
    }
    const float f0 = sc * W[kb * 1024 + v0 * 32 + w];
    const float f1 = sc * W[kb * 1024 + (v0 + 1) * 32 + w];
    return pack_bf16(f0, f1);
}

__global__ void __launch_bounds__(256)
prep_weights(const float* __restrict__ w000, const float* __restrict__ w011,
             const float* __restrict__ w101, const float* __restrict__ w110,
             const float* __restrict__ w111, unsigned int* __restrict__ bws)
{
    const int idx = blockIdx.x * 256 + threadIdx.x;
    if (idx < B_DWORDS)
        bws[idx] = weight_dword(idx, w000, w011, w101, w110, w111);
}

// ---------------- main kernel ----------------
// 512 threads = 16 waves; LDS = 320 KB (all 5 tensors, both w-halves).
// Every A fragment feeds two WMMAs (w-half 0/1); A-gen is packed bf16 VALU.
__global__ void __launch_bounds__(512)
tp_wmma_kernel(const float* __restrict__ x1, const float* __restrict__ x2,
               const float* __restrict__ w000, const float* __restrict__ w011,
               const float* __restrict__ w101, const float* __restrict__ w110,
               const float* __restrict__ w111,
               const unsigned int* __restrict__ bws, int use_ws,
               float* __restrict__ out, int ntiles)
{
    extern __shared__ unsigned int lds[];   // 81920 dwords = 320 KB

    const int tid = threadIdx.x;

    if (use_ws) {
        const v4u* __restrict__ src = (const v4u*)bws;
        v4u* dst = (v4u*)lds;
        for (int i = tid; i < B_DWORDS / 4; i += 512)
            dst[i] = src[i];
    } else {
        for (int idx = tid; idx < B_DWORDS; idx += 512)
            lds[idx] = weight_dword(idx, w000, w011, w101, w110, w111);
    }
    __syncthreads();

    const int wave = tid >> 5;
    const int lane = tid & 31;
    const int half = lane >> 4;
    const int n16  = lane & 15;
    const int wavesTotal = gridDim.x * 16;

    const v8f z8 = {0, 0, 0, 0, 0, 0, 0, 0};

    for (int tile = blockIdx.x * 16 + wave; tile < ntiles; tile += wavesTotal) {
        const int z = tile * 16 + n16;                 // A-fragment row M = lane%16
        const float* __restrict__ x1row = x1 + (size_t)z * 128;
        const float* __restrict__ x2row = x2 + (size_t)z * 128;

        // resident x2 fragments, packed bf16, A-matrix per-lane K pattern
        Frag x2s, x2va, x2vb, x2vc;
#pragma unroll
        for (int p = 0; p < 8; ++p) {
            const int v0 = ((p < 4) ? 0 : 16) + (half ? 8 : 0) + 2 * (p & 3);
            const int v1 = v0 + 1;
            x2s.w[p]  = cvt_pk(x2row[v0], x2row[v1]);
            x2va.w[p] = cvt_pk(x2row[32 + v0 * 3 + 0], x2row[32 + v1 * 3 + 0]);
            x2vb.w[p] = cvt_pk(x2row[32 + v0 * 3 + 1], x2row[32 + v1 * 3 + 1]);
            x2vc.w[p] = cvt_pk(x2row[32 + v0 * 3 + 2], x2row[32 + v1 * 3 + 2]);
        }

        // ================= pass 1: out0 and out1 =================
        v8f c00 = z8, c01 = z8;                      // out0, w-half 0/1
        v8f a10 = z8, a11 = z8, a12 = z8;            // out1 j=0..2, half0
        v8f b10 = z8, b11 = z8, b12 = z8;            // out1 j=0..2, half1

        float s0 = x1row[0];
        float sx = x1row[32], sy = x1row[33], sz = x1row[34];
#pragma unroll 1
        for (int kb = 0; kb < 32; ++kb) {
            const int kn = (kb + 1) & 31;            // pipelined scalar preload
            const float t0 = x1row[kn];
            const float tx = x1row[32 + kn * 3 + 0];
            const float ty = x1row[32 + kn * 3 + 1];
            const float tz = x1row[32 + kn * 3 + 2];

            const unsigned int u0 = splat_bf16(s0);
            const unsigned int ux = splat_bf16(sx);
            const unsigned int uy = splat_bf16(sy);
            const unsigned int uz = splat_bf16(sz);

            const unsigned int* bb = lds + kb * 256 + lane * 8;
            v16bf Bh0, Bh1, A;

            Bh0 = load_frag_lds(bb);                 // w000, half0
            Bh1 = load_frag_lds(bb + 8192);          // w000, half1
            A = mkfrag(x2s, u0);
            c00 = wmma_bf16(A, Bh0, c00);
            c01 = wmma_bf16(A, Bh1, c01);

            Bh0 = load_frag_lds(bb + 16384);         // w110
            Bh1 = load_frag_lds(bb + 24576);
            A = mkfrag(x2va, ux); c00 = wmma_bf16(A, Bh0, c00); c01 = wmma_bf16(A, Bh1, c01);
            A = mkfrag(x2vb, uy); c00 = wmma_bf16(A, Bh0, c00); c01 = wmma_bf16(A, Bh1, c01);
            A = mkfrag(x2vc, uz); c00 = wmma_bf16(A, Bh0, c00); c01 = wmma_bf16(A, Bh1, c01);

            Bh0 = load_frag_lds(bb + 32768);         // w011
            Bh1 = load_frag_lds(bb + 40960);
            A = mkfrag(x2va, u0); a10 = wmma_bf16(A, Bh0, a10); b10 = wmma_bf16(A, Bh1, b10);
            A = mkfrag(x2vb, u0); a11 = wmma_bf16(A, Bh0, a11); b11 = wmma_bf16(A, Bh1, b11);
            A = mkfrag(x2vc, u0); a12 = wmma_bf16(A, Bh0, a12); b12 = wmma_bf16(A, Bh1, b12);

            Bh0 = load_frag_lds(bb + 49152);         // w101
            Bh1 = load_frag_lds(bb + 57344);
            A = mkfrag(x2s, ux); a10 = wmma_bf16(A, Bh0, a10); b10 = wmma_bf16(A, Bh1, b10);
            A = mkfrag(x2s, uy); a11 = wmma_bf16(A, Bh0, a11); b11 = wmma_bf16(A, Bh1, b11);
            A = mkfrag(x2s, uz); a12 = wmma_bf16(A, Bh0, a12); b12 = wmma_bf16(A, Bh1, b12);

            s0 = t0; sx = tx; sy = ty; sz = tz;
        }

        {
            const int wg0 = n16, wg1 = 16 + n16;
#pragma unroll
            for (int r = 0; r < 8; ++r) {
                const int zg = tile * 16 + r + (half ? 8 : 0);
                const size_t o = (size_t)zg * 224;
                out[o + wg0]              = c00[r];
                out[o + wg1]              = c01[r];
                out[o + 32 + wg0 * 3 + 0] = a10[r];
                out[o + 32 + wg0 * 3 + 1] = a11[r];
                out[o + 32 + wg0 * 3 + 2] = a12[r];
                out[o + 32 + wg1 * 3 + 0] = b10[r];
                out[o + 32 + wg1 * 3 + 1] = b11[r];
                out[o + 32 + wg1 * 3 + 2] = b12[r];
            }
        }

        // ================= pass 2: out2 (antisymmetric l=1 path) =================
        v8f d20 = z8, d21 = z8, d22 = z8;            // half0
        v8f e20 = z8, e21 = z8, e22 = z8;            // half1

        sx = x1row[32]; sy = x1row[33]; sz = x1row[34];
#pragma unroll 1
        for (int kb = 0; kb < 32; ++kb) {
            const int kn = (kb + 1) & 31;
            const float tx = x1row[32 + kn * 3 + 0];
            const float ty = x1row[32 + kn * 3 + 1];
            const float tz = x1row[32 + kn * 3 + 2];

            const unsigned int ux = splat_bf16(sx);
            const unsigned int uy = splat_bf16(sy);
            const unsigned int uz = splat_bf16(sz);
            const unsigned int nx = splat_bf16(-sx);
            const unsigned int ny = splat_bf16(-sy);
            const unsigned int nz = splat_bf16(-sz);

            const unsigned int* bb = lds + kb * 256 + lane * 8;
            const v16bf Bh0 = load_frag_lds(bb + 65536);   // w111, half0
            const v16bf Bh1 = load_frag_lds(bb + 73728);   // w111, half1
            v16bf A;

            A = mkfrag2(x2vc, uy, x2vb, nz);  // comp (i=1,j=2)
            d20 = wmma_bf16(A, Bh0, d20); e20 = wmma_bf16(A, Bh1, e20);
            A = mkfrag2(x2va, uz, x2vc, nx);  // comp (i=2,j=0)
            d21 = wmma_bf16(A, Bh0, d21); e21 = wmma_bf16(A, Bh1, e21);
            A = mkfrag2(x2vb, ux, x2va, ny);  // comp (i=0,j=1)
            d22 = wmma_bf16(A, Bh0, d22); e22 = wmma_bf16(A, Bh1, e22);

            sx = tx; sy = ty; sz = tz;
        }

        {
            const int wg0 = n16, wg1 = 16 + n16;
#pragma unroll
            for (int r = 0; r < 8; ++r) {
                const int zg = tile * 16 + r + (half ? 8 : 0);
                const size_t o = (size_t)zg * 224;
                out[o + 128 + wg0 * 3 + 0] = d20[r];
                out[o + 128 + wg0 * 3 + 1] = d21[r];
                out[o + 128 + wg0 * 3 + 2] = d22[r];
                out[o + 128 + wg1 * 3 + 0] = e20[r];
                out[o + 128 + wg1 * 3 + 1] = e21[r];
                out[o + 128 + wg1 * 3 + 2] = e22[r];
            }
        }
    }
}

extern "C" void kernel_launch(void* const* d_in, const int* in_sizes, int n_in,
                              void* d_out, int out_size, void* d_ws, size_t ws_size,
                              hipStream_t stream) {
    const float* x1   = (const float*)d_in[0];
    const float* x2   = (const float*)d_in[1];
    const float* w000 = (const float*)d_in[2];
    const float* w011 = (const float*)d_in[3];
    const float* w101 = (const float*)d_in[4];
    const float* w110 = (const float*)d_in[5];
    const float* w111 = (const float*)d_in[6];

    const int n      = in_sizes[0] / 128;
    const int ntiles = n / 16;

    const int use_ws = (ws_size >= (size_t)B_DWORDS * 4) ? 1 : 0;
    unsigned int* bws = (unsigned int*)d_ws;
    if (use_ws) {
        prep_weights<<<B_DWORDS / 256, 256, 0, stream>>>(w000, w011, w101, w110,
                                                         w111, bws);
    }

    const size_t shmem = (size_t)B_DWORDS * 4;   // 320 KB dynamic LDS
    (void)hipFuncSetAttribute((const void*)tp_wmma_kernel,
                              hipFuncAttributeMaxDynamicSharedMemorySize, (int)shmem);

    int gx = (ntiles + 15) / 16;
    if (gx > 256) gx = 256;
    if (gx < 1)  gx = 1;
    dim3 grid(gx), block(512);
    tp_wmma_kernel<<<grid, block, shmem, stream>>>(x1, x2, w000, w011, w101, w110,
                                                   w111, bws, use_ws,
                                                   (float*)d_out, ntiles);
}
